// CustomMultiheadSelfAttention_21715354648993
// MI455X (gfx1250) — compile-verified, matrix operand order checked by reading, not run on hardware
//
#include <hip/hip_runtime.h>
#include <hip/hip_bf16.h>

#define DM 1024
#define SQ 2048
#define NH 16
#define DK 64

typedef unsigned short u16;
typedef __attribute__((ext_vector_type(16))) __bf16 v16bf;
typedef __attribute__((ext_vector_type(8)))  float  v8f;
typedef __attribute__((ext_vector_type(8)))  unsigned int v8u;

#define WMMA_BF16(a, b, c) \
  __builtin_amdgcn_wmma_f32_16x16x32_bf16(false, (a), false, (b), (short)0, (c), false, false)

// float -> bf16, round to nearest even
__device__ __forceinline__ u16 f2bf(float f) {
  unsigned int u = __float_as_uint(f);
  u += 0x7FFFu + ((u >> 16) & 1u);
  return (u16)(u >> 16);
}

// Load one 16-bit WMMA operand slice for this lane.
// Per ISA layout: elements 0..7 are 8 contiguous bf16 at p, elements 8..15 at p+16.
// Caller passes p = base + row*ld + k0 + half*8  (byte offset is 16B aligned).
__device__ __forceinline__ v16bf ld_op(const u16* p) {
  uint4 a = *(const uint4*)(p);
  uint4 b = *(const uint4*)(p + 16);
  v8u r;
  r[0] = a.x; r[1] = a.y; r[2] = a.z; r[3] = a.w;
  r[4] = b.x; r[5] = b.y; r[6] = b.z; r[7] = b.w;
  return __builtin_bit_cast(v16bf, r);
}

// ---------------------------------------------------------------- convert f32 -> bf16
__global__ void k_cvt_bf16(const float* __restrict__ in, u16* __restrict__ out, int n) {
  int i = blockIdx.x * 256 + threadIdx.x;
  if (i < n) out[i] = f2bf(in[i]);
}

// ---------------------------------------------------------------- NT GEMM: C[M,N] = A[M,K] * B[N,K]^T
// A, B bf16 row-major; C f32 row-major. 8 waves, each 32x32 tile; block tile 64x128.
__global__ __launch_bounds__(256) void k_gemm_nt(const u16* __restrict__ A,
                                                 const u16* __restrict__ B,
                                                 float* __restrict__ C,
                                                 int M, int N, int K) {
  const int lane = threadIdx.x & 31;
  const int wv   = threadIdx.x >> 5;
  const int half = lane >> 4;
  const int r    = lane & 15;
  const int m0 = blockIdx.x * 64  + (wv >> 2) * 32;
  const int n0 = blockIdx.y * 128 + (wv & 3) * 32;

  v8f acc[2][2] = {};
  for (int k0 = 0; k0 < K; k0 += 32) {
    v16bf a0 = ld_op(A + (size_t)(m0 + r)      * K + k0 + half * 8);
    v16bf a1 = ld_op(A + (size_t)(m0 + 16 + r) * K + k0 + half * 8);
    v16bf b0 = ld_op(B + (size_t)(n0 + r)      * K + k0 + half * 8);
    v16bf b1 = ld_op(B + (size_t)(n0 + 16 + r) * K + k0 + half * 8);
    acc[0][0] = WMMA_BF16(a0, b0, acc[0][0]);
    acc[0][1] = WMMA_BF16(a0, b1, acc[0][1]);
    acc[1][0] = WMMA_BF16(a1, b0, acc[1][0]);
    acc[1][1] = WMMA_BF16(a1, b1, acc[1][1]);
  }
#pragma unroll
  for (int mt = 0; mt < 2; ++mt)
#pragma unroll
    for (int nt = 0; nt < 2; ++nt)
#pragma unroll
      for (int i = 0; i < 8; ++i) {
        int row = m0 + mt * 16 + i + 8 * half;   // C/D layout: M = vgpr + 8*(lane>=16)
        int col = n0 + nt * 16 + r;              // N = lane & 15
        C[(size_t)row * N + col] = acc[mt][nt][i];
      }
}

// ---------------------------------------------------------------- RoPE + head scatter
// C: [B*S, DM] f32 projection result; dst: [B,H,S,DK] bf16 with RoPE applied.
__global__ void k_rope_scatter(const float* __restrict__ C, const int* __restrict__ pos,
                               u16* __restrict__ dst) {
  int idx = blockIdx.x * 256 + threadIdx.x;           // B*S*H*(DK/2) = 2^21 threads
  int i = idx & 31;
  int h = (idx >> 5) & 15;
  int s = (idx >> 9) & 2047;
  int b = idx >> 20;
  int d0 = i * 2;
  float inv = __expf((float)d0 * -0.14391156831212787f);   // 10000^(-d0/64)
  float ang = (float)pos[s] * inv;
  float sn, cs;
  __sincosf(ang, &sn, &cs);
  const float* src = C + ((size_t)(b * SQ + s)) * DM + h * DK + d0;
  float x0 = src[0], x1 = src[1];
  size_t o = (((size_t)(b * NH + h)) * SQ + s) * DK + d0;
  dst[o]     = f2bf(x0 * cs - x1 * sn);
  dst[o + 1] = f2bf(x1 * cs + x0 * sn);
}

// V scatter: C [B*S, DM] f32 -> Vt [B,H,DK,S] bf16 (transposed so PV is NT GEMM)
__global__ void k_v_scatter(const float* __restrict__ C, u16* __restrict__ Vt) {
  int idx = blockIdx.x * 256 + threadIdx.x;           // 4096*1024
  int col = idx & 1023;
  int rf  = idx >> 10;
  int s = rf & 2047;
  int b = rf >> 11;
  int h = col >> 6;
  int d = col & 63;
  Vt[((size_t)(b * NH + h) * DK + d) * SQ + s] = f2bf(C[idx]);
}

// ---------------------------------------------------------------- flash attention
// Q,K: [B,H,S,DK] bf16 (RoPE applied); Vt: [B,H,DK,S] bf16; O: [B*S, DM] bf16.
// 4 waves/block, each wave owns 16 query rows; loop over 32-key tiles (causal).
__global__ __launch_bounds__(128) void k_flash(const u16* __restrict__ Q,
                                               const u16* __restrict__ K,
                                               const u16* __restrict__ Vt,
                                               u16* __restrict__ O) {
  __shared__ __align__(16) u16 ldsP[4][16][40];       // per-wave P tile, padded pitch

  const int lane = threadIdx.x & 31;
  const int wv   = threadIdx.x >> 5;
  const int half = lane >> 4;
  const int r    = lane & 15;

  const int bh = blockIdx.y;                          // b*16 + h
  const int b  = bh >> 4;
  const int h  = bh & 15;
  const int qBase = blockIdx.x * 64 + wv * 16;

  const u16* Qh = Q  + (size_t)bh * SQ * DK;
  const u16* Kh = K  + (size_t)bh * SQ * DK;
  const u16* Vh = Vt + (size_t)bh * DK * SQ;

  // Q operand tiles (K-dim = d, two slices of 32)
  v16bf qa[2];
#pragma unroll
  for (int dk = 0; dk < 2; ++dk)
    qa[dk] = ld_op(Qh + (size_t)(qBase + r) * DK + dk * 32 + half * 8);

  v8f Oacc[4] = {};
  float mrow[8], lrow[8];
#pragma unroll
  for (int i = 0; i < 8; ++i) { mrow[i] = -3.0e30f; lrow[i] = 0.0f; }

  const float scale = 0.125f;                         // 1/sqrt(64)
  const int ktMax = (qBase + 15) >> 5;                // inclusive

  for (int kt = 0; kt <= ktMax; ++kt) {
    const int kGB = kt * 32;

    // scores S = Q * K^T for this 16x32 tile
    v16bf kb00 = ld_op(Kh + (size_t)(kGB + r)      * DK + 0  + half * 8);
    v16bf kb01 = ld_op(Kh + (size_t)(kGB + r)      * DK + 32 + half * 8);
    v16bf kb10 = ld_op(Kh + (size_t)(kGB + 16 + r) * DK + 0  + half * 8);
    v16bf kb11 = ld_op(Kh + (size_t)(kGB + 16 + r) * DK + 32 + half * 8);
    v8f z = {};
    v8f s0 = WMMA_BF16(qa[0], kb00, z);
    s0     = WMMA_BF16(qa[1], kb01, s0);
    v8f s1 = WMMA_BF16(qa[0], kb10, z);
    s1     = WMMA_BF16(qa[1], kb11, s1);

    // online softmax per row (row i+8*half lives in this 16-lane half at vgpr i)
#pragma unroll
    for (int i = 0; i < 8; ++i) {
      const int rowg = qBase + i + 8 * half;
      const int col0 = kGB + r;
      const int col1 = col0 + 16;
      float v0 = (col0 <= rowg) ? s0[i] * scale : -3.0e30f;
      float v1 = (col1 <= rowg) ? s1[i] * scale : -3.0e30f;
      float mx = fmaxf(v0, v1);
#pragma unroll
      for (int d = 1; d < 16; d <<= 1) mx = fmaxf(mx, __shfl_xor(mx, d, 32));
      float mnew = fmaxf(mrow[i], mx);
      float corr = __expf(mrow[i] - mnew);
      float p0 = __expf(v0 - mnew);
      float p1 = __expf(v1 - mnew);
      float ps = p0 + p1;
#pragma unroll
      for (int d = 1; d < 16; d <<= 1) ps += __shfl_xor(ps, d, 32);
      lrow[i] = lrow[i] * corr + ps;
      mrow[i] = mnew;
      Oacc[0][i] *= corr; Oacc[1][i] *= corr; Oacc[2][i] *= corr; Oacc[3][i] *= corr;
      // stage P (bf16) into LDS in (row, col) form
      ldsP[wv][i + 8 * half][r]      = f2bf(p0);
      ldsP[wv][i + 8 * half][16 + r] = f2bf(p1);
    }
    asm volatile("s_wait_dscnt 0" ::: "memory");      // wave-local LDS visibility

    // reload P in A-operand layout: row = lane&15, K slices by half
    v16bf pa;
    {
      const u16* pr = &ldsP[wv][r][0];
      uint4 a = *(const uint4*)(pr + half * 8);
      uint4 c = *(const uint4*)(pr + 16 + half * 8);
      v8u rr;
      rr[0] = a.x; rr[1] = a.y; rr[2] = a.z; rr[3] = a.w;
      rr[4] = c.x; rr[5] = c.y; rr[6] = c.z; rr[7] = c.w;
      pa = __builtin_bit_cast(v16bf, rr);
    }

    // O += P * V   (Vt rows = d, contiguous along keys -> NT operand)
#pragma unroll
    for (int nt = 0; nt < 4; ++nt) {
      v16bf vb = ld_op(Vh + (size_t)(nt * 16 + r) * SQ + kGB + half * 8);
      Oacc[nt] = WMMA_BF16(pa, vb, Oacc[nt]);
    }
  }

  // normalize and store O as bf16 [B*S, DM] row-major
#pragma unroll
  for (int nt = 0; nt < 4; ++nt)
#pragma unroll
    for (int i = 0; i < 8; ++i) {
      int row = qBase + i + 8 * half;
      int col = h * DK + nt * 16 + r;
      O[(size_t)(b * SQ + row) * DM + col] = f2bf(Oacc[nt][i] / lrow[i]);
    }
}

// ---------------------------------------------------------------- launcher
extern "C" void kernel_launch(void* const* d_in, const int* in_sizes, int n_in,
                              void* d_out, int out_size, void* d_ws, size_t ws_size,
                              hipStream_t stream) {
  const float* x   = (const float*)d_in[0];
  const int*   pos = (const int*)d_in[1];
  const float* Wq  = (const float*)d_in[2];
  const float* Wk  = (const float*)d_in[3];
  const float* Wv  = (const float*)d_in[4];
  const float* Wo  = (const float*)d_in[5];
  float* out = (float*)d_out;

  char* ws = (char*)d_ws;
  size_t off = 0;
  const size_t nX = (size_t)2 * SQ * DM;       // 4,194,304
  const size_t nW = (size_t)DM * DM;           // 1,048,576

  u16* Xbf = (u16*)(ws + off); off += nX * 2;
  u16* Wqb = (u16*)(ws + off); off += nW * 2;
  u16* Wkb = (u16*)(ws + off); off += nW * 2;
  u16* Wvb = (u16*)(ws + off); off += nW * 2;
  u16* Wob = (u16*)(ws + off); off += nW * 2;
  float* Ctmp = (float*)(ws + off); off += nX * 4;
  u16* Qb  = (u16*)(ws + off); off += nX * 2;
  u16* Kb  = (u16*)(ws + off); off += nX * 2;
  u16* Vtb = (u16*)(ws + off); off += nX * 2;
  u16* Ob  = (u16*)(ws + off); off += nX * 2;

  // 1) convert inputs to bf16
  k_cvt_bf16<<<(int)((nX + 255) / 256), 256, 0, stream>>>(x, Xbf, (int)nX);
  k_cvt_bf16<<<(int)((nW + 255) / 256), 256, 0, stream>>>(Wq, Wqb, (int)nW);
  k_cvt_bf16<<<(int)((nW + 255) / 256), 256, 0, stream>>>(Wk, Wkb, (int)nW);
  k_cvt_bf16<<<(int)((nW + 255) / 256), 256, 0, stream>>>(Wv, Wvb, (int)nW);
  k_cvt_bf16<<<(int)((nW + 255) / 256), 256, 0, stream>>>(Wo, Wob, (int)nW);

  dim3 gg(4096 / 64, DM / 128);   // (64, 8)
  const int ropeThreads = 2 * SQ * NH * (DK / 2);   // 2^21

  // 2) Q = x Wq^T -> RoPE -> [b,h,s,d] bf16
  k_gemm_nt<<<gg, 256, 0, stream>>>(Xbf, Wqb, Ctmp, 4096, DM, DM);
  k_rope_scatter<<<ropeThreads / 256, 256, 0, stream>>>(Ctmp, pos, Qb);
  // 3) K
  k_gemm_nt<<<gg, 256, 0, stream>>>(Xbf, Wkb, Ctmp, 4096, DM, DM);
  k_rope_scatter<<<ropeThreads / 256, 256, 0, stream>>>(Ctmp, pos, Kb);
  // 4) V (stored transposed [b,h,d,s])
  k_gemm_nt<<<gg, 256, 0, stream>>>(Xbf, Wvb, Ctmp, 4096, DM, DM);
  k_v_scatter<<<(int)(nX / 256), 256, 0, stream>>>(Ctmp, Vtb);

  // 5) causal flash attention
  dim3 fg(SQ / 64, 2 * NH);       // (32, 32)
  k_flash<<<fg, 128, 0, stream>>>(Qb, Kb, Vtb, Ob);

  // 6) final projection: out = O Wo^T (f32 out)
  k_gemm_nt<<<gg, 256, 0, stream>>>(Ob, Wob, out, 4096, DM, DM);
}
